// SoftAggONNX_ori_20134806683722
// MI455X (gfx1250) — matrix operand order repaired and checked by reading, not compiled, not run on hardware
//
#include <hip/hip_runtime.h>
#include <hip/hip_bf16.h>

// CDNA5 (gfx1250) wave32 WMMA implementation of SoftAgg:
//   fx = x@Wf+bf ; e = exp(x@Wg+bg) ; denom = segsum(e, ix)
//   y  = segsum(fx * e/denom[ix], ix) @ Wh + bh ; out = y[ix]
//
// GEMMs use v_wmma_f32_16x16x32_bf16 (fp32 accumulate, bf16 operands).
// Memory-bound workload: x (256MB) is read twice (denominator pass +
// recompute pass) instead of materializing 512MB of [N,C] intermediates.
// [G,C] accumulators (16MB) stay L2-resident and take f32 global atomics.

typedef __attribute__((ext_vector_type(16))) __bf16 v16bf;
typedef __attribute__((ext_vector_type(8)))  float  v8f;

#define SOFTAGG_G 4096   // number of groups (fixed by the reference problem)

// ---------------------------------------------------------------------------
// A-fragment (16x32, bf16) loaded directly from an fp32 row-major matrix.
// ISA layout (CDNA5 ISA, 16-bit A 16x32):
//   lane m   (m<16): row m, K = {kb+0..7,  kb+16..23}
//   lane m+16      : row m, K = {kb+8..15, kb+24..31}
// v16bf element j < 8 -> first chunk, j >= 8 -> second chunk (sequential K).
// ---------------------------------------------------------------------------
static __device__ __forceinline__ v16bf load_a_frag_f32(const float* __restrict__ X,
                                                        int row0, int kb, int stride) {
  const int lane = threadIdx.x & 31;
  const int m    = lane & 15;
  const int hi   = lane >> 4;
  const float4* p = reinterpret_cast<const float4*>(
      X + (size_t)(row0 + m) * (size_t)stride + kb + (hi << 3));
  float4 c0 = p[0];   // K = base+0..3
  float4 c1 = p[1];   // K = base+4..7
  float4 c2 = p[4];   // K = base+16..19
  float4 c3 = p[5];   // K = base+20..23
  v16bf a;
  a[0]  = (__bf16)c0.x; a[1]  = (__bf16)c0.y; a[2]  = (__bf16)c0.z; a[3]  = (__bf16)c0.w;
  a[4]  = (__bf16)c1.x; a[5]  = (__bf16)c1.y; a[6]  = (__bf16)c1.z; a[7]  = (__bf16)c1.w;
  a[8]  = (__bf16)c2.x; a[9]  = (__bf16)c2.y; a[10] = (__bf16)c2.z; a[11] = (__bf16)c2.w;
  a[12] = (__bf16)c3.x; a[13] = (__bf16)c3.y; a[14] = (__bf16)c3.z; a[15] = (__bf16)c3.w;
  return a;
}

// Packed B fragments: per (colTile, kStep) block, 32 lanes x 16 bf16, lane-contiguous.
static __device__ __forceinline__ v16bf load_b_frag(const __bf16* __restrict__ Wp,
                                                    int tileIdx) {
  const int lane = threadIdx.x & 31;
  return reinterpret_cast<const v16bf*>(Wp)[(size_t)tileIdx * 32 + lane];
}

// ---------------------------------------------------------------------------
// Pack fp32 weight W[C,C] (row-major, out[n] = sum_k x[k]*W[k*C+n]) into the
// CDNA5 B-fragment layout (32x16 bf16 per tile):
//   lane n<16 : col n, K = {kb+0..7,  kb+16..23}
//   lane n+16 : col n, K = {kb+8..15, kb+24..31}
// ---------------------------------------------------------------------------
__global__ void pack_w_kernel(const float* __restrict__ W, __bf16* __restrict__ Wp, int C) {
  const int ksteps = C >> 5;
  const int bid  = blockIdx.x;            // = nt * ksteps + kt
  const int kt   = bid % ksteps;
  const int nt   = bid / ksteps;
  const int lane = threadIdx.x;           // 32 threads per block
  const int n    = (nt << 4) + (lane & 15);
  const int kb   = (kt << 5) + ((lane >> 4) << 3);
  __bf16* dst = Wp + ((size_t)bid * 32 + lane) * 16;
#pragma unroll
  for (int j = 0; j < 16; ++j)
    dst[j] = (__bf16)W[(size_t)(kb + j + ((j >> 3) << 3)) * (size_t)C + n];
}

__global__ void zero_kernel(float* __restrict__ p, size_t n4) {
  size_t i = (size_t)blockIdx.x * blockDim.x + threadIdx.x;
  if (i < n4)
    reinterpret_cast<float4*>(p)[i] = make_float4(0.f, 0.f, 0.f, 0.f);
}

// ---------------------------------------------------------------------------
// Pass A: gx = x@Wg (WMMA), e = exp(gx + bg), denom[ix[row]][col] += e
// One wave per 16x16 output tile; 8 waves (same row tile) per block for
// A-fragment reuse through the cache hierarchy.
// ---------------------------------------------------------------------------
__global__ void __launch_bounds__(256)
passA_kernel(const float* __restrict__ x, const int* __restrict__ ix,
             const __bf16* __restrict__ WgP, const float* __restrict__ biasG,
             float* __restrict__ denom, int C) {
  const int ksteps   = C >> 5;
  const int colTiles = C >> 4;
  const int wave = blockIdx.x * (blockDim.x >> 5) + (threadIdx.x >> 5);
  const int rt = wave / colTiles;
  const int ct = wave % colTiles;
  const int lane = threadIdx.x & 31;

  v8f acc = {};
#pragma unroll 4
  for (int kt = 0; kt < ksteps; ++kt) {
    v16bf a = load_a_frag_f32(x, rt << 4, kt << 5, C);
    v16bf b = load_b_frag(WgP, ct * ksteps + kt);
    acc = __builtin_amdgcn_wmma_f32_16x16x32_bf16(false, a, false, b,
                                                  (short)0, acc, false, false);
  }

  const int col = (ct << 4) + (lane & 15);
  const int hi  = lane >> 4;
  const float bgc = biasG[col];
#pragma unroll
  for (int j = 0; j < 8; ++j) {
    const int row = (rt << 4) + j + (hi << 3);   // C/D layout: VGPR j, lane half
    const float e = __expf(acc[j] + bgc);
    const int g = ix[row];
    atomicAdd(denom + (size_t)g * (size_t)C + col, e);
  }
}

// ---------------------------------------------------------------------------
// Pass B: recompute fx and gx (shared A fragment -> two WMMAs per k-step),
// w = exp(gx+bg)/denom[g][col], yacc[g][col] += (fx+bf)*w
// ---------------------------------------------------------------------------
__global__ void __launch_bounds__(256)
passB_kernel(const float* __restrict__ x, const int* __restrict__ ix,
             const __bf16* __restrict__ WfP, const float* __restrict__ biasF,
             const __bf16* __restrict__ WgP, const float* __restrict__ biasG,
             const float* __restrict__ denom, float* __restrict__ yacc, int C) {
  const int ksteps   = C >> 5;
  const int colTiles = C >> 4;
  const int wave = blockIdx.x * (blockDim.x >> 5) + (threadIdx.x >> 5);
  const int rt = wave / colTiles;
  const int ct = wave % colTiles;
  const int lane = threadIdx.x & 31;

  v8f accF = {};
  v8f accG = {};
#pragma unroll 4
  for (int kt = 0; kt < ksteps; ++kt) {
    v16bf a  = load_a_frag_f32(x, rt << 4, kt << 5, C);
    v16bf bF = load_b_frag(WfP, ct * ksteps + kt);
    v16bf bG = load_b_frag(WgP, ct * ksteps + kt);
    accF = __builtin_amdgcn_wmma_f32_16x16x32_bf16(false, a, false, bF,
                                                   (short)0, accF, false, false);
    accG = __builtin_amdgcn_wmma_f32_16x16x32_bf16(false, a, false, bG,
                                                   (short)0, accG, false, false);
  }

  const int col = (ct << 4) + (lane & 15);
  const int hi  = lane >> 4;
  const float bfc = biasF[col];
  const float bgc = biasG[col];
#pragma unroll
  for (int j = 0; j < 8; ++j) {
    const int row = (rt << 4) + j + (hi << 3);
    const int g = ix[row];
    const size_t o = (size_t)g * (size_t)C + col;
    const float w = __expf(accG[j] + bgc) / denom[o];
    atomicAdd(yacc + o, (accF[j] + bfc) * w);
  }
}

// ---------------------------------------------------------------------------
// yout = yacc @ Wh + bh   ([G,C] x [C,C])
// ---------------------------------------------------------------------------
__global__ void __launch_bounds__(256)
out_gemm_kernel(const float* __restrict__ yacc, const __bf16* __restrict__ WhP,
                const float* __restrict__ biasH, float* __restrict__ yout, int C) {
  const int ksteps   = C >> 5;
  const int colTiles = C >> 4;
  const int wave = blockIdx.x * (blockDim.x >> 5) + (threadIdx.x >> 5);
  const int rt = wave / colTiles;
  const int ct = wave % colTiles;
  const int lane = threadIdx.x & 31;

  v8f acc = {};
#pragma unroll 4
  for (int kt = 0; kt < ksteps; ++kt) {
    v16bf a = load_a_frag_f32(yacc, rt << 4, kt << 5, C);
    v16bf b = load_b_frag(WhP, ct * ksteps + kt);
    acc = __builtin_amdgcn_wmma_f32_16x16x32_bf16(false, a, false, b,
                                                  (short)0, acc, false, false);
  }

  const int col = (ct << 4) + (lane & 15);
  const int hi  = lane >> 4;
  const float bhc = biasH[col];
#pragma unroll
  for (int j = 0; j < 8; ++j) {
    const int row = (rt << 4) + j + (hi << 3);
    yout[(size_t)row * (size_t)C + col] = acc[j] + bhc;
  }
}

// out[i,:] = yout[ix[i],:]   (float4-vectorized gather)
__global__ void gather_kernel(const float* __restrict__ yout, const int* __restrict__ ix,
                              float* __restrict__ out, int C, int N) {
  const size_t c4 = (size_t)(C >> 2);
  size_t t = (size_t)blockIdx.x * blockDim.x + threadIdx.x;
  if (t >= (size_t)N * c4) return;
  const size_t i = t / c4;
  const size_t c = (t % c4) << 2;
  const int g = ix[i];
  reinterpret_cast<float4*>(out + i * (size_t)C + c)[0] =
      reinterpret_cast<const float4*>(yout + (size_t)g * (size_t)C + c)[0];
}

extern "C" void kernel_launch(void* const* d_in, const int* in_sizes, int n_in,
                              void* d_out, int out_size, void* d_ws, size_t ws_size,
                              hipStream_t stream) {
  const float* x  = (const float*)d_in[0];
  const int*   ix = (const int*)d_in[1];
  const float* Wf = (const float*)d_in[2];
  const float* bf = (const float*)d_in[3];
  const float* Wg = (const float*)d_in[4];
  const float* bg = (const float*)d_in[5];
  const float* Wh = (const float*)d_in[6];
  const float* bh = (const float*)d_in[7];

  const int N = in_sizes[1];   // number of elements (length of ix)
  const int C = in_sizes[3];   // feature dim (length of bf)
  const int G = SOFTAGG_G;     // number of groups

  // Workspace layout (all naturally aligned; ~25.5 MB for the reference shape)
  float*  denom = (float*)d_ws;                        // [G,C]
  float*  yacc  = denom + (size_t)G * C;               // [G,C]
  float*  yout  = yacc  + (size_t)G * C;               // [G,C]
  __bf16* WfP   = (__bf16*)(yout + (size_t)G * C);     // C*C bf16, packed B frags
  const size_t wElems = (size_t)C * (size_t)C;
  __bf16* WgP   = WfP + wElems;
  __bf16* WhP   = WgP + wElems;
  (void)ws_size; (void)n_in; (void)out_size;

  // 1) zero the accumulators (denom + yacc, contiguous)
  {
    size_t n4 = ((size_t)G * C * 2) >> 2;
    zero_kernel<<<(unsigned)((n4 + 255) / 256), 256, 0, stream>>>(denom, n4);
  }

  // 2) pack weights into WMMA B-fragment layout (one wave per 16x32 tile)
  {
    const int wtiles = (C >> 4) * (C >> 5);
    pack_w_kernel<<<wtiles, 32, 0, stream>>>(Wf, WfP, C);
    pack_w_kernel<<<wtiles, 32, 0, stream>>>(Wg, WgP, C);
    pack_w_kernel<<<wtiles, 32, 0, stream>>>(Wh, WhP, C);
  }

  // 3) pass A: denominator accumulation
  {
    const int waves = (N >> 4) * (C >> 4);      // one wave per 16x16 tile
    passA_kernel<<<waves / 8, 256, 0, stream>>>(x, ix, WgP, bg, denom, C);
  }

  // 4) pass B: weighted aggregation
  {
    const int waves = (N >> 4) * (C >> 4);
    passB_kernel<<<waves / 8, 256, 0, stream>>>(x, ix, WfP, bf, WgP, bg,
                                                denom, yacc, C);
  }

  // 5) output projection y@Wh + bh
  {
    const int waves = (G >> 4) * (C >> 4);
    out_gemm_kernel<<<waves / 8, 256, 0, stream>>>(yacc, WhP, bh, yout, C);
  }

  // 6) gather back to elements
  {
    size_t threads = (size_t)N * (size_t)(C >> 2);
    gather_kernel<<<(unsigned)((threads + 255) / 256), 256, 0, stream>>>(
        yout, ix, (float*)d_out, C, N);
  }
}
